// LayoutLMv2ForTokenClassification_46359876993345
// MI455X (gfx1250) — matrix operand order repaired and checked by reading, not compiled, not run on hardware
//
#include <hip/hip_runtime.h>
#include <hip/hip_bf16.h>
#include <cstddef>
#include <cstdint>

// Problem constants (bert-base-uncased token-classification head with neighbor fusion)
#define V_SIZE 30522
#define B_SIZE 2
#define S_SIZE 512
#define D_SIZE 768
#define J_SIZE 12
#define K_NBR  4
#define NLAB   7
#define M_ROWS (B_SIZE * S_SIZE)          // 1024
#define KBIG   (13 * D_SIZE)              // 9984
#define NBIG   (J_SIZE * D_SIZE)          // 9216
#define K2     (2 * D_SIZE)               // 1536
#define N2     (D_SIZE)                   // 768

typedef __attribute__((ext_vector_type(16))) _Float16 v16h;
typedef __attribute__((ext_vector_type(8)))  float    v8f;

union HF16 { v16h v; uint4 q[2]; };

// ---------------------------------------------------------------------------
// Dictionary build kernels (replicate the persistent dict semantics)
// ---------------------------------------------------------------------------
__global__ void init_marks(int* present, int* lastpos, int n) {
    int i = blockIdx.x * blockDim.x + threadIdx.x;
    if (i < n) { present[i] = 0; lastpos[i] = -1; }
}

__global__ void reset_lastpos(int* lastpos, int n) {
    int i = blockIdx.x * blockDim.x + threadIdx.x;
    if (i < n) lastpos[i] = -1;
}

__global__ void scatter_last(const int* __restrict__ ids, int* lastpos) {
    int s = blockIdx.x * blockDim.x + threadIdx.x;
    if (s < S_SIZE) atomicMax(&lastpos[ids[s]], s);
}

// grid = S blocks, 192 threads (float4 covers 768 floats)
__global__ void update_table(const int* __restrict__ ids,
                             const int* __restrict__ lastpos,
                             const float* __restrict__ seq_b,   // [S,D]
                             float* __restrict__ table,         // [V,D]
                             int* __restrict__ present) {
    int s  = blockIdx.x;
    int id = ids[s];
    if (lastpos[id] != s) return;          // only last occurrence writes
    int t = threadIdx.x;
    const float4* src = reinterpret_cast<const float4*>(seq_b + (size_t)s * D_SIZE);
    float4*       dst = reinterpret_cast<float4*>(table + (size_t)id * D_SIZE);
    dst[t] = src[t];
    if (t == 0) present[id] = 1;
}

// grid = S*J blocks, 192 threads; fn_b[s,j,:] = sum_k present ? table[nb] : 0
__global__ void gather_fn(const int* __restrict__ nb_b,          // [S,J,K]
                          const float* __restrict__ table,
                          const int* __restrict__ present,
                          float* __restrict__ fn_b) {            // [S,J,D]
    int sj = blockIdx.x;
    int t  = threadIdx.x;
    const int* nbp = nb_b + (size_t)sj * K_NBR;
    float4 acc = make_float4(0.f, 0.f, 0.f, 0.f);
    #pragma unroll
    for (int k = 0; k < K_NBR; ++k) {
        int id = nbp[k];
        if (present[id]) {
            float4 v = *reinterpret_cast<const float4*>(table + (size_t)id * D_SIZE + t * 4);
            acc.x += v.x; acc.y += v.y; acc.z += v.z; acc.w += v.w;
        }
    }
    reinterpret_cast<float4*>(fn_b)[(size_t)sj * 192 + t] = acc;
}

// ---------------------------------------------------------------------------
// Pack rij = [seq ; n_flat] as f16  [M, 13D]
// ---------------------------------------------------------------------------
__global__ void pack_rij(const float* __restrict__ seq,   // [M,768]
                         const float* __restrict__ fn,    // [M,9216]
                         _Float16* __restrict__ rij) {    // [M,9984]
    int idx = blockIdx.x * blockDim.x + threadIdx.x;      // one float4 each
    int m   = idx / (KBIG / 4);
    int col = (idx % (KBIG / 4)) * 4;
    float4 v;
    if (col < D_SIZE)
        v = *reinterpret_cast<const float4*>(seq + (size_t)m * D_SIZE + col);
    else
        v = *reinterpret_cast<const float4*>(fn + (size_t)m * NBIG + (col - D_SIZE));
    _Float16* dst = rij + (size_t)m * KBIG + col;
    dst[0] = (_Float16)v.x; dst[1] = (_Float16)v.y;
    dst[2] = (_Float16)v.z; dst[3] = (_Float16)v.w;
}

// ---------------------------------------------------------------------------
// WMMA GEMM: Cout[m,n] = epilogue( sum_k A[m,k] * W[n,k] + bias[n] )
//   MODE 0: out = sigmoid(val) * gate[m,n]   (gated MLP -> c)
//   MODE 1: out = tanh(val)                  (attention projection)
// A is f16 [M,K] row-major (staged via async global->LDS); W is f32 [N,K]
// row-major (register-staged, converted to f16 into LDS).
// Tile 128x128x32, double-buffered LDS, 256 threads = 8 waves (2M x 4N),
// wave tile 64x32 -> 8 v_wmma_f32_16x16x32_f16 per K-step.
// ---------------------------------------------------------------------------
#define BM 128
#define BN 128
#define BK 32
#define LDP 40   // padded row length in halves (80B, 16B-aligned rows)

template <int MODE>
__global__ __launch_bounds__(256)
void wmma_gemm_f16(const _Float16* __restrict__ A,
                   const float* __restrict__ W,
                   const float* __restrict__ bias,
                   const float* __restrict__ gate,
                   float* __restrict__ Cout,
                   int M, int N, int K) {
    __shared__ __align__(16) _Float16 As[2][BM][LDP];
    __shared__ __align__(16) _Float16 Bs[2][BN][LDP];

    const int tid   = threadIdx.x;
    const int wid   = tid >> 5;
    const int lane  = tid & 31;
    const int waveM = wid >> 2;   // 0..1
    const int waveN = wid & 3;    // 0..3
    const int m0 = blockIdx.y * BM;
    const int n0 = blockIdx.x * BN;

    v8f acc[4][2];
    #pragma unroll
    for (int mt = 0; mt < 4; ++mt)
        #pragma unroll
        for (int nt = 0; nt < 2; ++nt)
            acc[mt][nt] = (v8f){0.f, 0.f, 0.f, 0.f, 0.f, 0.f, 0.f, 0.f};

    const int mlo   = lane & 15;
    const int khalf = lane >> 4;

    // --- staging helpers -----------------------------------------------
    // A tile: 128x32 f16, 512 x 16B chunks, 2 per thread, async global->LDS
    auto stageA_async = [&](int buf, int kk) {
        #pragma unroll
        for (int i = 0; i < 2; ++i) {
            int c  = tid + i * 256;        // 0..511
            int r  = c >> 2;
            int kc = (c & 3) * 8;
            unsigned lds = (unsigned)(uintptr_t)&As[buf][r][kc];
            unsigned long long gp =
                (unsigned long long)(uintptr_t)(A + (size_t)(m0 + r) * K + kk + kc);
            asm volatile("global_load_async_to_lds_b128 %0, %1, off"
                         :: "v"(lds), "v"(gp) : "memory");
        }
    };
    // W tile: 128x32 f32, 4 independent float4 loads per thread
    auto stageW_load = [&](float4* wreg, int kk) {
        #pragma unroll
        for (int i = 0; i < 4; ++i) {
            int c  = tid + i * 256;        // 0..1023
            int r  = c >> 3;
            int kc = (c & 7) * 4;
            wreg[i] = *reinterpret_cast<const float4*>(
                W + (size_t)(n0 + r) * K + kk + kc);
        }
    };
    auto stageW_store = [&](int buf, const float4* wreg) {
        #pragma unroll
        for (int i = 0; i < 4; ++i) {
            int c  = tid + i * 256;
            int r  = c >> 3;
            int kc = (c & 7) * 4;
            _Float16* dst = &Bs[buf][r][kc];
            dst[0] = (_Float16)wreg[i].x; dst[1] = (_Float16)wreg[i].y;
            dst[2] = (_Float16)wreg[i].z; dst[3] = (_Float16)wreg[i].w;
        }
    };

    // --- prologue: stage tile 0 into buffer 0 ---------------------------
    {
        stageA_async(0, 0);
        float4 wreg[4];
        stageW_load(wreg, 0);
        stageW_store(0, wreg);
        asm volatile("s_wait_asynccnt 0x0" ::: "memory");
        __syncthreads();
    }

    const int KT = K / BK;
    for (int kt = 0; kt < KT; ++kt) {
        const int cur  = kt & 1;
        const int nxt  = cur ^ 1;
        const bool more = (kt + 1) < KT;

        float4 wreg[4];
        if (more) {
            stageA_async(nxt, (kt + 1) * BK);   // async, overlaps compute
            stageW_load(wreg, (kt + 1) * BK);   // 4 loads in flight
        }

        // fragments per ISA layout:
        //  A 16x32: lanes 0-15 -> K 0..7 / 16..23 ; lanes 16-31 -> K 8..15 / 24..31
        //  B 32x16: lanes 0-15 -> K 0..7 / 16..23 ; lanes 16-31 -> K 8..15 / 24..31
        HF16 afr[4], bfr[2];
        #pragma unroll
        for (int mt = 0; mt < 4; ++mt) {
            const _Float16* ap = &As[cur][waveM * 64 + mt * 16 + mlo][0];
            afr[mt].q[0] = *reinterpret_cast<const uint4*>(ap + khalf * 8);
            afr[mt].q[1] = *reinterpret_cast<const uint4*>(ap + 16 + khalf * 8);
        }
        #pragma unroll
        for (int nt = 0; nt < 2; ++nt) {
            const _Float16* bp = &Bs[cur][waveN * 32 + nt * 16 + mlo][0] + khalf * 8;
            bfr[nt].q[0] = *reinterpret_cast<const uint4*>(bp);
            bfr[nt].q[1] = *reinterpret_cast<const uint4*>(bp + 16);
        }
        #pragma unroll
        for (int mt = 0; mt < 4; ++mt)
            #pragma unroll
            for (int nt = 0; nt < 2; ++nt)
                acc[mt][nt] = __builtin_amdgcn_wmma_f32_16x16x32_f16(
                    false, afr[mt].v, false, bfr[nt].v,
                    (short)0, acc[mt][nt], false, false);

        if (more) stageW_store(nxt, wreg);      // convert+store next W tile

        asm volatile("s_wait_asynccnt 0x0" ::: "memory");
        __syncthreads();
    }

    // epilogue: C/D layout -> lane: n = lane&15, row = (lane>>4)*8 + v
    #pragma unroll
    for (int mt = 0; mt < 4; ++mt) {
        #pragma unroll
        for (int nt = 0; nt < 2; ++nt) {
            int col   = n0 + waveN * 32 + nt * 16 + (lane & 15);
            int rbase = m0 + waveM * 64 + mt * 16 + (lane >> 4) * 8;
            float bcol = bias[col];
            #pragma unroll
            for (int v = 0; v < 8; ++v) {
                int row = rbase + v;
                float val = acc[mt][nt][v] + bcol;
                size_t o = (size_t)row * N + col;
                if (MODE == 0) {
                    float g = 1.0f / (1.0f + __expf(-val));
                    Cout[o] = g * gate[o];
                } else {
                    Cout[o] = tanhf(val);
                }
            }
        }
    }
}

// ---------------------------------------------------------------------------
// Per-token 12-way attention: scores -> softmax -> mix; emit [mix ; seq] f16
// 384 threads = 12 waves (one per neighbor slot j)
// ---------------------------------------------------------------------------
__global__ __launch_bounds__(384)
void attn_mix(const float* __restrict__ seq,    // [M,768]
              const float* __restrict__ cmat,   // [M,12,768]
              _Float16* __restrict__ comb) {    // [M,1536]
    int m    = blockIdx.x;
    int tid  = threadIdx.x;
    int wid  = tid >> 5;
    int lane = tid & 31;
    __shared__ float sc[J_SIZE];

    const float* srow = seq  + (size_t)m * D_SIZE;
    const float* crow = cmat + (size_t)m * NBIG;

    {   // score_j = dot(seq_row, c_row_j)
        const float* cj = crow + wid * D_SIZE;
        float p = 0.f;
        for (int d = lane; d < D_SIZE; d += 32) p += srow[d] * cj[d];
        p += __shfl_xor(p, 16, 32); p += __shfl_xor(p, 8, 32);
        p += __shfl_xor(p, 4, 32);  p += __shfl_xor(p, 2, 32);
        p += __shfl_xor(p, 1, 32);
        if (lane == 0) sc[wid] = p;
    }
    __syncthreads();

    float w[J_SIZE];
    float mx = sc[0];
    #pragma unroll
    for (int j = 1; j < J_SIZE; ++j) mx = fmaxf(mx, sc[j]);
    float sum = 0.f;
    #pragma unroll
    for (int j = 0; j < J_SIZE; ++j) { w[j] = __expf(sc[j] - mx); sum += w[j]; }
    float inv = 1.0f / sum;
    #pragma unroll
    for (int j = 0; j < J_SIZE; ++j) w[j] *= inv;

    for (int d = tid; d < D_SIZE; d += 384) {
        float mix = 0.f;
        #pragma unroll
        for (int j = 0; j < J_SIZE; ++j) mix += w[j] * crow[j * D_SIZE + d];
        comb[(size_t)m * K2 + d]          = (_Float16)mix;
        comb[(size_t)m * K2 + D_SIZE + d] = (_Float16)srow[d];
    }
}

// ---------------------------------------------------------------------------
// Final classifier: logits[m,l] = b[l] + seq.W1 + attn_out.W2   (tiny)
// ---------------------------------------------------------------------------
__global__ void cls_kernel(const float* __restrict__ seq,
                           const float* __restrict__ ao,
                           const float* __restrict__ Wc,   // [7,1536]
                           const float* __restrict__ bc,
                           float* __restrict__ out) {
    int idx = blockIdx.x * blockDim.x + threadIdx.x;
    if (idx >= M_ROWS * NLAB) return;
    int m = idx / NLAB, l = idx % NLAB;
    const float* s  = seq + (size_t)m * D_SIZE;
    const float* a  = ao  + (size_t)m * D_SIZE;
    const float* w1 = Wc  + (size_t)l * K2;
    const float* w2 = w1 + D_SIZE;
    float acc = bc[l];
    for (int d = 0; d < D_SIZE; ++d) acc += s[d] * w1[d] + a[d] * w2[d];
    out[idx] = acc;
}

// ---------------------------------------------------------------------------
// Host orchestration
// ---------------------------------------------------------------------------
static inline size_t align256(size_t x) { return (x + 255) & ~(size_t)255; }

extern "C" void kernel_launch(void* const* d_in, const int* in_sizes, int n_in,
                              void* d_out, int out_size, void* d_ws, size_t ws_size,
                              hipStream_t stream) {
    (void)in_sizes; (void)n_in; (void)out_size; (void)ws_size;
    const float* seq   = (const float*)d_in[0];   // [B,S,D]
    const int*   ids   = (const int*)d_in[1];     // [B,S]
    const int*   nbr   = (const int*)d_in[2];     // [B,S,J,K]
    const float* Wmlp  = (const float*)d_in[3];   // [9216,9984]
    const float* bmlp  = (const float*)d_in[4];   // [9216]
    const float* Wattn = (const float*)d_in[5];   // [768,1536]
    const float* battn = (const float*)d_in[6];   // [768]
    const float* Wcls  = (const float*)d_in[7];   // [7,1536]
    const float* bcls  = (const float*)d_in[8];   // [7]
    float*       out   = (float*)d_out;           // [B,S,7]

    // workspace carve-up
    char* ws = (char*)d_ws;
    size_t off = 0;
    float* table = (float*)(ws + off); off = align256(off + (size_t)V_SIZE * D_SIZE * 4);
    int* present = (int*)(ws + off);   off = align256(off + (size_t)V_SIZE * 4);
    int* lastpos = (int*)(ws + off);   off = align256(off + (size_t)V_SIZE * 4);
    float* fn    = (float*)(ws + off); off = align256(off + (size_t)M_ROWS * NBIG * 4);
    _Float16* rij = (_Float16*)(ws + off); off = align256(off + (size_t)M_ROWS * KBIG * 2);
    float* cmat  = (float*)(ws + off); off = align256(off + (size_t)M_ROWS * NBIG * 4);
    _Float16* comb = (_Float16*)(ws + off); off = align256(off + (size_t)M_ROWS * K2 * 2);
    float* aout  = (float*)(ws + off); off = align256(off + (size_t)M_ROWS * N2 * 4);

    // 1) dictionary build + gather, sequential over batch (dict persists)
    init_marks<<<(V_SIZE + 255) / 256, 256, 0, stream>>>(present, lastpos, V_SIZE);
    for (int b = 0; b < B_SIZE; ++b) {
        const int*   ids_b = ids + (size_t)b * S_SIZE;
        const float* seq_b = seq + (size_t)b * S_SIZE * D_SIZE;
        const int*   nb_b  = nbr + (size_t)b * S_SIZE * J_SIZE * K_NBR;
        float*       fn_b  = fn + (size_t)b * S_SIZE * J_SIZE * D_SIZE;
        scatter_last<<<(S_SIZE + 255) / 256, 256, 0, stream>>>(ids_b, lastpos);
        update_table<<<S_SIZE, 192, 0, stream>>>(ids_b, lastpos, seq_b, table, present);
        gather_fn<<<S_SIZE * J_SIZE, 192, 0, stream>>>(nb_b, table, present, fn_b);
        if (b + 1 < B_SIZE)
            reset_lastpos<<<(V_SIZE + 255) / 256, 256, 0, stream>>>(lastpos, V_SIZE);
    }

    // 2) pack rij = [seq ; fn] as f16
    pack_rij<<<(M_ROWS * (KBIG / 4)) / 256, 256, 0, stream>>>(seq, fn, rij);

    // 3) big gated-MLP GEMM with fused sigmoid*gate epilogue -> cmat
    wmma_gemm_f16<0><<<dim3(NBIG / BN, M_ROWS / BM), 256, 0, stream>>>(
        rij, Wmlp, bmlp, fn, cmat, M_ROWS, NBIG, KBIG);

    // 4) per-token attention + combined pack (f16)
    attn_mix<<<M_ROWS, 384, 0, stream>>>(seq, cmat, comb);

    // 5) attention projection GEMM with tanh epilogue -> aout
    wmma_gemm_f16<1><<<dim3(N2 / BN, M_ROWS / BM), 256, 0, stream>>>(
        comb, Wattn, battn, nullptr, aout, M_ROWS, N2, K2);

    // 6) classifier
    cls_kernel<<<(M_ROWS * NLAB + 255) / 256, 256, 0, stream>>>(seq, aout, Wcls, bcls, out);
}
